// MetaGAT_53017076302393
// MI455X (gfx1250) — matrix operand order repaired and checked by reading, not compiled, not run on hardware
//
#include <hip/hip_runtime.h>

#define N_NODES 20000
#define E_EDGES 320000
#define ETOT    (E_EDGES + N_NODES)   // 340000 with self loops
#define IN_DIM  128
#define HIDDEN  256
#define HEADS1  4
#define HD1     64
#define OUT_DIM 128
#define NEG_SLOPE 0.2f
#define BN_EPS    1e-5f
#define NEG_INF (-__builtin_inff())

typedef __attribute__((ext_vector_type(16))) __bf16 v16bf;
typedef __attribute__((ext_vector_type(8)))  float  v8f;

// ---------------------------------------------------------------- fill
__global__ void fill_kernel(float* __restrict__ p, float v, int n) {
  int i = blockIdx.x * blockDim.x + threadIdx.x;
  if (i < n) p[i] = v;
}

// ---------------------------------------------------------------- operand packing
// Pack A[M,K] f32 -> bf16 fragments in the ISA 16-bit A layout.
// Element t = ((mt*KB + kblk)*32 + lane); each lane owns 16 halves (32B).
__global__ void pack_a_kernel(const float* __restrict__ A, __bf16* __restrict__ pA,
                              int Mtiles, int KB, int K) {
  int t = blockIdx.x * blockDim.x + threadIdx.x;
  if (t >= Mtiles * KB * 32) return;
  int lane = t & 31;
  int w    = t >> 5;
  int kblk = w % KB;
  int mt   = w / KB;
  int row  = mt * 16 + (lane & 15);
  int ka   = (lane >> 4) * 8;
  const float* ar = A + (size_t)row * K + kblk * 32 + ka;
  v16bf v;
#pragma unroll
  for (int i = 0; i < 8; ++i) {
    v[i]     = (__bf16)ar[i];        // K = kblk*32 + ka + i
    v[i + 8] = (__bf16)ar[16 + i];   // K = kblk*32 + ka + 16 + i
  }
  *(v16bf*)(pA + (size_t)t * 16) = v;
}

// Pack B[K,N] f32 -> bf16 fragments in the ISA 16-bit B layout.
// Element t = ((kblk*NT + nt)*32 + lane); each lane owns 16 halves (32B).
__global__ void pack_b_kernel(const float* __restrict__ B, __bf16* __restrict__ pB,
                              int KB, int NT, int N) {
  int t = blockIdx.x * blockDim.x + threadIdx.x;
  if (t >= KB * NT * 32) return;
  int lane  = t & 31;
  int w     = t >> 5;
  int nt    = w % NT;
  int kblk  = w / NT;
  int col   = nt * 16 + (lane & 15);
  int kbase = kblk * 32 + (lane >> 4) * 8;
  v16bf v;
#pragma unroll
  for (int i = 0; i < 16; ++i) v[i] = (__bf16)B[(size_t)(kbase + (i & 7) + ((i >> 3) << 4)) * N + col];
  *(v16bf*)(pB + (size_t)t * 16) = v;
}

// ---------------------------------------------------------------- WMMA GEMM (packed)
// C[M,N] = A x B from pre-packed bf16 fragments. One wave computes a 16x64
// strip (4 n-tiles), reusing the A fragment 4x. KB = K/32 (fully unrolled).
template<int KB>
__global__ void wmma_gemm_packed_kernel(const __bf16* __restrict__ pA,
                                        const __bf16* __restrict__ pB,
                                        float* __restrict__ C,
                                        int Mtiles, int NT, int Ncols) {
  const int wave = (blockIdx.x * blockDim.x + threadIdx.x) >> 5;
  const int lane = threadIdx.x & 31;
  const int ngrp = NT >> 2;                      // groups of 4 n-tiles
  const int mt = wave / ngrp;
  const int ng = wave - mt * ngrp;
  if (mt >= Mtiles) return;                      // whole-wave uniform exit

  v8f acc0 = {0.f,0.f,0.f,0.f,0.f,0.f,0.f,0.f};
  v8f acc1 = acc0, acc2 = acc0, acc3 = acc0;
  const v16bf* aP = (const v16bf*)pA + (size_t)mt * KB * 32 + lane;
  const v16bf* bP = (const v16bf*)pB + (size_t)(ng * 4) * 32 + lane;
#pragma unroll
  for (int kblk = 0; kblk < KB; ++kblk) {
    v16bf a = aP[(size_t)kblk * 32];
    const v16bf* bk = bP + (size_t)kblk * NT * 32;
    v16bf b0 = bk[0];
    v16bf b1 = bk[32];
    v16bf b2 = bk[64];
    v16bf b3 = bk[96];
    acc0 = __builtin_amdgcn_wmma_f32_16x16x32_bf16(false, a, false, b0, (short)0, acc0, false, false);
    acc1 = __builtin_amdgcn_wmma_f32_16x16x32_bf16(false, a, false, b1, (short)0, acc1, false, false);
    acc2 = __builtin_amdgcn_wmma_f32_16x16x32_bf16(false, a, false, b2, (short)0, acc2, false, false);
    acc3 = __builtin_amdgcn_wmma_f32_16x16x32_bf16(false, a, false, b3, (short)0, acc3, false, false);
  }
  const int l16 = lane & 15;
  const int lh8 = (lane >> 4) * 8;
#pragma unroll
  for (int j = 0; j < 4; ++j) {
    v8f acc = (j == 0) ? acc0 : (j == 1) ? acc1 : (j == 2) ? acc2 : acc3;
    float* crow = C + (size_t)(mt * 16 + lh8) * Ncols + (size_t)(ng * 4 + j) * 16 + l16;
#pragma unroll
    for (int i = 0; i < 8; ++i) crow[(size_t)i * Ncols] = acc[i];
  }
}

// ---------------------------------------------------------------- attention coeffs
__global__ void attn_coeff_kernel(const float* __restrict__ h,
                                  const float* __restrict__ att_s,
                                  const float* __restrict__ att_d,
                                  float* __restrict__ as, float* __restrict__ ad,
                                  int n_nodes, int heads, int hd) {
  const int gw   = (blockIdx.x * blockDim.x + threadIdx.x) >> 5;
  const int lane = threadIdx.x & 31;
  if (gw >= n_nodes * heads) return;
  const int node = gw / heads;
  const int hh   = gw - node * heads;
  const float* rowp = h + (size_t)node * heads * hd + (size_t)hh * hd;
  float s = 0.f, d = 0.f;
  for (int i = lane; i < hd; i += 32) {
    float v = rowp[i];
    s += v * att_s[hh * hd + i];
    d += v * att_d[hh * hd + i];
  }
#pragma unroll
  for (int m = 16; m > 0; m >>= 1) {
    s += __shfl_xor(s, m, 32);
    d += __shfl_xor(d, m, 32);
  }
  if (lane == 0) { as[gw] = s; ad[gw] = d; }
}

// ---------------------------------------------------------------- float atomic max
__device__ __forceinline__ void atomicMaxFloat(float* addr, float v) {
  if (v >= 0.f) atomicMax((int*)addr, __float_as_int(v));
  else          atomicMin((unsigned int*)addr, __float_as_uint(v));
}

// ---------------------------------------------------------------- edge phase 1
__global__ void edge_scores_kernel(const int* __restrict__ ei, int Eorig, int Etot,
                                   const float* __restrict__ as,
                                   const float* __restrict__ ad,
                                   float* __restrict__ eval,
                                   float* __restrict__ mx, int heads) {
  int idx = blockIdx.x * blockDim.x + threadIdx.x;
  if (idx >= Etot * heads) return;
  int e  = idx / heads;
  int hh = idx - e * heads;
  int s, d;
  if (e < Eorig) { s = ei[e]; d = ei[Eorig + e]; }
  else           { s = d = e - Eorig; }
  float v = as[s * heads + hh] + ad[d * heads + hh];
  v = (v > 0.f) ? v : NEG_SLOPE * v;
  eval[idx] = v;
  atomicMaxFloat(&mx[d * heads + hh], v);
}

// ---------------------------------------------------------------- edge phase 2
__global__ void edge_exp_kernel(const int* __restrict__ ei, int Eorig, int Etot,
                                float* __restrict__ eval,
                                const float* __restrict__ mx,
                                float* __restrict__ den, int heads) {
  int idx = blockIdx.x * blockDim.x + threadIdx.x;
  if (idx >= Etot * heads) return;
  int e  = idx / heads;
  int hh = idx - e * heads;
  int d  = (e < Eorig) ? ei[Eorig + e] : (e - Eorig);
  float ex = expf(eval[idx] - mx[d * heads + hh]);
  eval[idx] = ex;
  atomicAdd(&den[d * heads + hh], ex);
}

// ---------------------------------------------------------------- edge phase 3
__global__ void edge_aggregate_kernel(const int* __restrict__ ei, int Eorig, int Etot,
                                      const float* __restrict__ hfeat,
                                      const float* __restrict__ ex,
                                      const float* __restrict__ den,
                                      float* __restrict__ outp,
                                      int heads, int hd) {
  const int e    = (blockIdx.x * blockDim.x + threadIdx.x) >> 5;
  const int lane = threadIdx.x & 31;
  if (e >= Etot) return;
  int s, d;
  if (e < Eorig) { s = ei[e]; d = ei[Eorig + e]; }
  else           { s = d = e - Eorig; }
  const int C = heads * hd;
  const float* src_row = hfeat + (size_t)s * C;
  float* dst_row = outp + (size_t)d * C;
  for (int c = lane; c < C; c += 32) {
    int hh = c / hd;
    float alpha = ex[e * heads + hh] / (den[d * heads + hh] + 1e-16f);
    atomicAdd(&dst_row[c], src_row[c] * alpha);
  }
}

// ---------------------------------------------------------------- BN reduce
__global__ void bn_reduce_kernel(const float* __restrict__ x,
                                 float* __restrict__ sums, float* __restrict__ sumsq,
                                 int n_rows, int C) {
  const int c   = threadIdx.x;
  const int rpb = (n_rows + gridDim.x - 1) / gridDim.x;
  const int r0  = blockIdx.x * rpb;
  const int r1  = (r0 + rpb < n_rows) ? (r0 + rpb) : n_rows;
  float s = 0.f, s2 = 0.f;
  for (int r = r0; r < r1; ++r) {
    float v = x[(size_t)r * C + c];
    s += v; s2 += v * v;
  }
  atomicAdd(&sums[c], s);
  atomicAdd(&sumsq[c], s2);
}

// ---------------------------------------------------------------- BN apply (+ELU)
__global__ void bn_apply_kernel(const float* __restrict__ x,
                                const float* __restrict__ sums,
                                const float* __restrict__ sumsq,
                                const float* __restrict__ gamma,
                                const float* __restrict__ beta,
                                float* __restrict__ y,
                                int n_rows, int C, int do_elu) {
  int idx = blockIdx.x * blockDim.x + threadIdx.x;
  if (idx >= n_rows * C) return;
  int c = idx % C;
  float inv_n = 1.f / (float)n_rows;
  float mean  = sums[c] * inv_n;
  float var   = sumsq[c] * inv_n - mean * mean;
  float v = (x[idx] - mean) * rsqrtf(var + BN_EPS) * gamma[c] + beta[c];
  if (do_elu) v = (v > 0.f) ? v : (expf(v) - 1.f);
  y[idx] = v;
}

// ---------------------------------------------------------------- launch
extern "C" void kernel_launch(void* const* d_in, const int* in_sizes, int n_in,
                              void* d_out, int out_size, void* d_ws, size_t ws_size,
                              hipStream_t stream) {
  const float* x      = (const float*)d_in[0];
  const int*   ei     = (const int*)d_in[1];   // [2, E]: row0=src, row1=dst
  const float* W1     = (const float*)d_in[2];
  const float* att_s1 = (const float*)d_in[3];
  const float* att_d1 = (const float*)d_in[4];
  // bias1 (d_in[5]) cancels under training-mode BN (per-column constant)
  const float* g1     = (const float*)d_in[6];
  const float* b1     = (const float*)d_in[7];
  const float* W2     = (const float*)d_in[8];
  const float* att_s2 = (const float*)d_in[9];
  const float* att_d2 = (const float*)d_in[10];
  // bias2 (d_in[11]) cancels under BN
  const float* g2     = (const float*)d_in[12];
  const float* b2     = (const float*)d_in[13];
  float* out = (float*)d_out;
  float* ws  = (float*)d_ws;

  // workspace layout (floats; packed regions are 32B aligned)
  float*  h1    = ws;                 // 5,120,000  (layer-1 linear; reused as feat2)
  float*  agg   = ws + 5120000;       // 5,120,000  (agg1; later h2 [2.56M] + agg2 [2.56M])
  float*  as    = ws + 10240000;      // 80,000 max
  float*  ad    = as + 80000;         // 80,000 max
  float*  mx    = ad + 80000;         // 80,000 max
  float*  den   = mx + 80000;         // 80,000 max
  float*  ex    = den + 80000;        // 1,360,000 max
  float*  sums  = ex + 1360000;       // 256
  float*  sumsq = sums + 256;         // 256
  __bf16* pA    = (__bf16*)(ws + 11920512);  // up to 5,120,000 bf16 (layer 2)
  __bf16* pB    = (__bf16*)(ws + 14480512);  // up to 32,768 bf16

  const int blk = 256;
  auto cdiv = [](int a, int b) { return (a + b - 1) / b; };
  const int Mtiles = N_NODES / 16;    // 1250

  // ================= Layer 1: GATConv(128 -> 4x64, concat) =================
  fill_kernel<<<cdiv(N_NODES*HEADS1, blk), blk, 0, stream>>>(mx,  NEG_INF, N_NODES*HEADS1);
  fill_kernel<<<cdiv(N_NODES*HEADS1, blk), blk, 0, stream>>>(den, 0.f,     N_NODES*HEADS1);
  fill_kernel<<<cdiv(N_NODES*HIDDEN, blk), blk, 0, stream>>>(agg, 0.f,     N_NODES*HIDDEN);
  fill_kernel<<<2, blk, 0, stream>>>(sums, 0.f, 512);

  {
    const int KB = IN_DIM / 32;       // 4
    const int NT = HIDDEN / 16;       // 16
    pack_a_kernel<<<cdiv(Mtiles*KB*32, blk), blk, 0, stream>>>(x,  pA, Mtiles, KB, IN_DIM);
    pack_b_kernel<<<cdiv(KB*NT*32,   blk), blk, 0, stream>>>(W1, pB, KB, NT, HIDDEN);
    int waves = Mtiles * (NT / 4);    // 5000
    wmma_gemm_packed_kernel<4><<<cdiv(waves*32, blk), blk, 0, stream>>>(pA, pB, h1, Mtiles, NT, HIDDEN);
  }
  attn_coeff_kernel<<<cdiv(N_NODES*HEADS1*32, blk), blk, 0, stream>>>(
      h1, att_s1, att_d1, as, ad, N_NODES, HEADS1, HD1);
  edge_scores_kernel<<<cdiv(ETOT*HEADS1, blk), blk, 0, stream>>>(
      ei, E_EDGES, ETOT, as, ad, ex, mx, HEADS1);
  edge_exp_kernel<<<cdiv(ETOT*HEADS1, blk), blk, 0, stream>>>(
      ei, E_EDGES, ETOT, ex, mx, den, HEADS1);
  edge_aggregate_kernel<<<cdiv(ETOT*32, blk), blk, 0, stream>>>(
      ei, E_EDGES, ETOT, h1, ex, den, agg, HEADS1, HD1);
  bn_reduce_kernel<<<200, HIDDEN, 0, stream>>>(agg, sums, sumsq, N_NODES, HIDDEN);
  bn_apply_kernel<<<cdiv(N_NODES*HIDDEN, blk), blk, 0, stream>>>(
      agg, sums, sumsq, g1, b1, /*feat2=*/h1, N_NODES, HIDDEN, /*elu=*/1);

  // ================= Layer 2: GATConv(256 -> 128, 1 head) ==================
  float* feat2 = h1;
  float* h2    = agg;                 // 2,560,000
  float* agg2  = agg + 2560000;       // 2,560,000

  fill_kernel<<<cdiv(N_NODES, blk), blk, 0, stream>>>(mx,  NEG_INF, N_NODES);
  fill_kernel<<<cdiv(N_NODES, blk), blk, 0, stream>>>(den, 0.f,     N_NODES);
  fill_kernel<<<cdiv(N_NODES*OUT_DIM, blk), blk, 0, stream>>>(agg2, 0.f, N_NODES*OUT_DIM);
  fill_kernel<<<2, blk, 0, stream>>>(sums, 0.f, 512);

  {
    const int KB = HIDDEN / 32;       // 8
    const int NT = OUT_DIM / 16;      // 8
    pack_a_kernel<<<cdiv(Mtiles*KB*32, blk), blk, 0, stream>>>(feat2, pA, Mtiles, KB, HIDDEN);
    pack_b_kernel<<<cdiv(KB*NT*32,   blk), blk, 0, stream>>>(W2, pB, KB, NT, OUT_DIM);
    int waves = Mtiles * (NT / 4);    // 2500
    wmma_gemm_packed_kernel<8><<<cdiv(waves*32, blk), blk, 0, stream>>>(pA, pB, h2, Mtiles, NT, OUT_DIM);
  }
  attn_coeff_kernel<<<cdiv(N_NODES*32, blk), blk, 0, stream>>>(
      h2, att_s2, att_d2, as, ad, N_NODES, 1, OUT_DIM);
  edge_scores_kernel<<<cdiv(ETOT, blk), blk, 0, stream>>>(
      ei, E_EDGES, ETOT, as, ad, ex, mx, 1);
  edge_exp_kernel<<<cdiv(ETOT, blk), blk, 0, stream>>>(
      ei, E_EDGES, ETOT, ex, mx, den, 1);
  edge_aggregate_kernel<<<cdiv(ETOT*32, blk), blk, 0, stream>>>(
      ei, E_EDGES, ETOT, h2, ex, den, agg2, 1, OUT_DIM);
  bn_reduce_kernel<<<200, OUT_DIM, 0, stream>>>(agg2, sums, sumsq, N_NODES, OUT_DIM);
  bn_apply_kernel<<<cdiv(N_NODES*OUT_DIM, blk), blk, 0, stream>>>(
      agg2, sums, sumsq, g2, b2, out, N_NODES, OUT_DIM, /*elu=*/0);
}
